// LogLinearAttention_55903294325378
// MI455X (gfx1250) — compile-verified, matrix operand order checked by reading, not run on hardware
//
#include <hip/hip_runtime.h>
#include <hip/hip_bf16.h>
#include <math.h>
#include <stdint.h>

// ---------------------------------------------------------------------------
// Types for CDNA5 (gfx1250) wave32 WMMA
// ---------------------------------------------------------------------------
typedef __bf16 bf16_t;
typedef __attribute__((ext_vector_type(16))) __bf16 v16bf;
typedef __attribute__((ext_vector_type(8)))  __bf16 v8bf;
typedef __attribute__((ext_vector_type(8)))  float  v8f;

#define WMMA_BF16(A, B, C) \
  __builtin_amdgcn_wmma_f32_16x16x32_bf16(false, (A), false, (B), (short)0, (C), false, false)

// --- gfx1250 async global->LDS path (guarded: falls back to sync copies) ---
#if defined(__has_builtin)
#if __has_builtin(__builtin_amdgcn_global_load_async_to_lds_b128) && \
    __has_builtin(__builtin_amdgcn_s_wait_asynccnt)
#define HAVE_ASYNC_LDS 1
#endif
#endif

#if defined(HAVE_ASYNC_LDS)
// Builtin signature (from hipcc diagnostics): arg0 = v4i AS(1)* (global src),
// arg1 = v4i AS(3)* (LDS dst), arg2/arg3 = constant int offset / cpol.
typedef int v4i_vs __attribute__((vector_size(16)));
typedef __attribute__((address_space(1))) v4i_vs as1_v4i;
typedef __attribute__((address_space(3))) v4i_vs as3_v4i;
// generic->AS1: numeric identity; generic LDS ptr low 32 bits == LDS offset.
#define AS1P(p) ((as1_v4i*)(uintptr_t)(p))
#define AS3P(p) ((as3_v4i*)(uintptr_t)(uint32_t)(uintptr_t)(p))
#define ASYNC_CP_B128(gp, lp) \
  __builtin_amdgcn_global_load_async_to_lds_b128(AS1P(gp), AS3P(lp), 0, 0)
#define ASYNC_WAIT(n) __builtin_amdgcn_s_wait_asynccnt(n)
#endif

__device__ __forceinline__ v8f zero8() {
  v8f z;
#pragma unroll
  for (int i = 0; i < 8; ++i) z[i] = 0.0f;
  return z;
}

// Load one 16x32 bf16 fragment (A layout; B mirrored, so the same loader works
// when the B-source matrix is stored [N,K] row-major).
// p must point at (row_base + fk) where fk = (lane>=16 ? 8 : 0) elements.
__device__ __forceinline__ v16bf frag_from(const bf16_t* p) {
  v8bf lo = *reinterpret_cast<const v8bf*>(p);
  v8bf hi = *reinterpret_cast<const v8bf*>(p + 16);
  v16bf f;
#pragma unroll
  for (int i = 0; i < 8; ++i) { f[i] = lo[i]; f[i + 8] = hi[i]; }
  return f;
}

__device__ __forceinline__ float half16_max(float v) {
  v = fmaxf(v, __shfl_xor(v, 1, 32));
  v = fmaxf(v, __shfl_xor(v, 2, 32));
  v = fmaxf(v, __shfl_xor(v, 4, 32));
  v = fmaxf(v, __shfl_xor(v, 8, 32));
  return v;
}
__device__ __forceinline__ float half16_sum(float v) {
  v += __shfl_xor(v, 1, 32);
  v += __shfl_xor(v, 2, 32);
  v += __shfl_xor(v, 4, 32);
  v += __shfl_xor(v, 8, 32);
  return v;
}

// ---------------------------------------------------------------------------
// Problem constants
// ---------------------------------------------------------------------------
#define BATCH 4
#define TLEN  4096
#define CDIM  1024
#define HEADS 16
#define DH    64
#define NCHUNK 8
#define CSIZE 512          // TLEN / NCHUNK
#define MROWS (BATCH * TLEN)

// ---------------------------------------------------------------------------
// fp32 -> bf16 conversion
// ---------------------------------------------------------------------------
__global__ __launch_bounds__(256) void cvt_f32_bf16_kernel(
    const float* __restrict__ in, bf16_t* __restrict__ out, int n) {
  int i = blockIdx.x * 256 + threadIdx.x;
  if (i < n) out[i] = (bf16_t)in[i];
}

// ---------------------------------------------------------------------------
// lambda gate MLP: lam[h][t] = sigmoid(Wl2 @ relu(log(t+1)*Wl1 + bl1) + bl2)
// ---------------------------------------------------------------------------
__global__ __launch_bounds__(256) void lambda_kernel(
    const float* __restrict__ Wl1, const float* __restrict__ bl1,
    const float* __restrict__ Wl2, const float* __restrict__ bl2,
    float* __restrict__ lam /* [HEADS][TLEN] */) {
  int t = blockIdx.x * 256 + threadIdx.x;
  if (t >= TLEN) return;
  float lp = __logf((float)t + 1.0f);
  float h1[64];
#pragma unroll
  for (int i = 0; i < 64; ++i) h1[i] = fmaxf(lp * Wl1[i] + bl1[i], 0.0f);
#pragma unroll
  for (int h = 0; h < HEADS; ++h) {
    float a = bl2[h];
#pragma unroll
    for (int i = 0; i < 64; ++i) a += Wl2[h * 64 + i] * h1[i];
    lam[h * TLEN + t] = 1.0f / (1.0f + __expf(-a));
  }
}

// ---------------------------------------------------------------------------
// WMMA GEMM: C[M,N] = A[M,K] @ B[N,K]^T   (A, B bf16 row-major; f32 accum)
// Block tile 128x128, 8 waves (2x4), wave tile 64x32 = 4x2 WMMA accumulators.
// Double-buffered LDS; tiles staged with GLOBAL_LOAD_ASYNC_TO_LDS_B128 when
// available (prefetch of tile k+1 overlaps compute of tile k).
// ---------------------------------------------------------------------------
#define GK 32
#define LPITCH 40   // 32 + 8 pad: conflict-free, 80B row pitch (16B aligned)

template <bool OUT_F32>
__global__ __launch_bounds__(256) void gemm_bf16_kernel(
    const bf16_t* __restrict__ A, const bf16_t* __restrict__ B,
    void* __restrict__ Cout, const float* __restrict__ bias,
    int M, int N, int K) {
  __shared__ __align__(16) bf16_t sA[2][128][LPITCH];
  __shared__ __align__(16) bf16_t sB[2][128][LPITCH];

  const int tid  = threadIdx.x;
  const int wave = tid >> 5;
  const int lane = tid & 31;
  const int wm   = wave >> 2;       // 0..1  -> 64 rows
  const int wn   = wave & 3;        // 0..3  -> 32 cols
  const int half = lane >> 4;
  const int frow = lane & 15;
  const int fk   = half << 3;       // 0 or 8 (element offset within k-chunk)
  const int bm   = blockIdx.x * 128;
  const int bn   = blockIdx.y * 128;

  v8f acc[4][2];
#pragma unroll
  for (int mi = 0; mi < 4; ++mi)
#pragma unroll
    for (int ni = 0; ni < 2; ++ni) acc[mi][ni] = zero8();

  const int r0 = tid >> 2;          // 0..63
  const int cc = (tid & 3) * 8;     // 0,8,16,24

  // Stage one 128x32 A-tile + B-tile into LDS buffer `bi` (4 x 16B per thread)
  auto stage = [&](int bi, int k0) {
#if defined(HAVE_ASYNC_LDS)
    ASYNC_CP_B128(&A[(size_t)(bm + r0) * K + k0 + cc],      &sA[bi][r0][cc]);
    ASYNC_CP_B128(&A[(size_t)(bm + r0 + 64) * K + k0 + cc], &sA[bi][r0 + 64][cc]);
    ASYNC_CP_B128(&B[(size_t)(bn + r0) * K + k0 + cc],      &sB[bi][r0][cc]);
    ASYNC_CP_B128(&B[(size_t)(bn + r0 + 64) * K + k0 + cc], &sB[bi][r0 + 64][cc]);
#else
    *reinterpret_cast<v8bf*>(&sA[bi][r0][cc]) =
        *reinterpret_cast<const v8bf*>(&A[(size_t)(bm + r0) * K + k0 + cc]);
    *reinterpret_cast<v8bf*>(&sA[bi][r0 + 64][cc]) =
        *reinterpret_cast<const v8bf*>(&A[(size_t)(bm + r0 + 64) * K + k0 + cc]);
    *reinterpret_cast<v8bf*>(&sB[bi][r0][cc]) =
        *reinterpret_cast<const v8bf*>(&B[(size_t)(bn + r0) * K + k0 + cc]);
    *reinterpret_cast<v8bf*>(&sB[bi][r0 + 64][cc]) =
        *reinterpret_cast<const v8bf*>(&B[(size_t)(bn + r0 + 64) * K + k0 + cc]);
#endif
  };

  int buf = 0;
  stage(0, 0);
  for (int k0 = 0; k0 < K; k0 += GK) {
    const bool more = (k0 + GK) < K;
    if (more) stage(buf ^ 1, k0 + GK);   // prefetch next tile
#if defined(HAVE_ASYNC_LDS)
    if (more) ASYNC_WAIT(4);             // current tile's 4 loads landed
    else      ASYNC_WAIT(0);
#endif
    __syncthreads();                     // all waves' current tile visible

    v16bf af[4], bfv[2];
#pragma unroll
    for (int mi = 0; mi < 4; ++mi)
      af[mi] = frag_from(&sA[buf][wm * 64 + mi * 16 + frow][fk]);
#pragma unroll
    for (int ni = 0; ni < 2; ++ni)
      bfv[ni] = frag_from(&sB[buf][wn * 32 + ni * 16 + frow][fk]);

#pragma unroll
    for (int mi = 0; mi < 4; ++mi)
#pragma unroll
      for (int ni = 0; ni < 2; ++ni)
        acc[mi][ni] = WMMA_BF16(af[mi], bfv[ni], acc[mi][ni]);

    __syncthreads();                     // done reading before buf is re-staged
    buf ^= 1;
  }

  // Write out. C/D layout: VGPR r, lane l -> (M = r + 8*half, N = lane&15)
#pragma unroll
  for (int mi = 0; mi < 4; ++mi) {
#pragma unroll
    for (int ni = 0; ni < 2; ++ni) {
      int gm0 = bm + wm * 64 + mi * 16 + 8 * half;
      int gn  = bn + wn * 32 + ni * 16 + frow;
      float bv = OUT_F32 ? bias[gn] : 0.0f;
#pragma unroll
      for (int r = 0; r < 8; ++r) {
        float v = acc[mi][ni][r];
        if (OUT_F32)
          reinterpret_cast<float*>(Cout)[(size_t)(gm0 + r) * N + gn] = v + bv;
        else
          reinterpret_cast<bf16_t*>(Cout)[(size_t)(gm0 + r) * N + gn] = (bf16_t)v;
      }
    }
  }
}

// ---------------------------------------------------------------------------
// Chunked gated attention (flash-style over the 512-key chunk).
// Block: 256 threads = 8 waves; block owns 128 q-rows of one (b,h,chunk);
// each wave owns 16 q-rows. Keys processed in blocks of 64.
// qkv: bf16 [B, T, 3*CDIM]; lam: f32 [HEADS][TLEN]; out: bf16 [B, T, CDIM]
// ---------------------------------------------------------------------------
#define APITCH 72   // 64 + 8 pad elements; 144B pitch (16B aligned, bank-clean)

__global__ __launch_bounds__(256) void attn_kernel(
    const bf16_t* __restrict__ qkv, const float* __restrict__ lam,
    bf16_t* __restrict__ outb) {
  __shared__ __align__(16) bf16_t sK[64][APITCH];        // [key][dh]
  __shared__ __align__(16) bf16_t sVt[64][APITCH];       // [dh][key]
  __shared__ __align__(16) bf16_t sP[8][16][APITCH];     // per-wave P tiles

  const int tid   = threadIdx.x;
  const int wave  = tid >> 5;
  const int lane  = tid & 31;
  const int half  = lane >> 4;
  const int frow  = lane & 15;
  const int fk    = half << 3;

  const int idx   = blockIdx.x;
  const int qt    = idx & 3;           // 4 q-tiles of 128 per chunk
  const int chunk = (idx >> 2) & 7;
  const int h     = (idx >> 5) & 15;
  const int b     = idx >> 9;

  const size_t rowstride = 3 * CDIM;
  const size_t base      = (size_t)b * TLEN * rowstride;
  const float  scale     = 0.125f;     // dh^-0.5 = 1/8

  // Q fragments for this wave's 16 rows (K = dh = 64 -> 2 k-steps)
  const int qrow = chunk * CSIZE + qt * 128 + wave * 16 + frow;
  const bf16_t* qp = qkv + base + (size_t)qrow * rowstride + h * DH;
  v16bf qf[2];
#pragma unroll
  for (int ks = 0; ks < 2; ++ks) qf[ks] = frag_from(qp + ks * 32 + fk);

  v8f o[4];
#pragma unroll
  for (int nt = 0; nt < 4; ++nt) o[nt] = zero8();
  v8f m_acc, l_acc;
#pragma unroll
  for (int i = 0; i < 8; ++i) { m_acc[i] = -1e30f; l_acc[i] = 0.0f; }

  for (int kb = 0; kb < 8; ++kb) {     // 8 key-blocks of 64
    __syncthreads();
    // Stage K tile [64 keys x 64 dh] (async when available) and V^T tile
    for (int c = tid; c < 512; c += 256) {
      int kr = c >> 3, kc = (c & 7) * 8;
      int kt = chunk * CSIZE + kb * 64 + kr;
      const bf16_t* src = qkv + base + (size_t)kt * rowstride + CDIM + h * DH + kc;
#if defined(HAVE_ASYNC_LDS)
      ASYNC_CP_B128(src, &sK[kr][kc]);
#else
      *reinterpret_cast<v8bf*>(&sK[kr][kc]) = *reinterpret_cast<const v8bf*>(src);
#endif
      const bf16_t* vsrc = qkv + base + (size_t)kt * rowstride + 2 * CDIM + h * DH + kc;
      v8bf vv = *reinterpret_cast<const v8bf*>(vsrc);
#pragma unroll
      for (int i = 0; i < 8; ++i) sVt[kc + i][kr] = vv[i];
    }
#if defined(HAVE_ASYNC_LDS)
    ASYNC_WAIT(0);
#endif
    __syncthreads();

    // S = Q @ K^T : 4 tiles of 16 keys, each 2 WMMA over dh
    v8f sc[4];
#pragma unroll
    for (int nt = 0; nt < 4; ++nt) {
      v8f s = zero8();
#pragma unroll
      for (int ks = 0; ks < 2; ++ks) {
        v16bf bfrag = frag_from(&sK[nt * 16 + frow][ks * 32 + fk]);
        s = WMMA_BF16(qf[ks], bfrag, s);
      }
      sc[nt] = s;
    }

    // Gate keys with scale * lam[h][key]  (key = lane&15 within each tile)
#pragma unroll
    for (int nt = 0; nt < 4; ++nt) {
      float g = scale * lam[h * TLEN + chunk * CSIZE + kb * 64 + nt * 16 + frow];
#pragma unroll
      for (int i = 0; i < 8; ++i) sc[nt][i] *= g;
    }

    // Online softmax: row = (i + 8*half); rows live in 16-lane halves
    v8f mx = sc[0];
#pragma unroll
    for (int nt = 1; nt < 4; ++nt)
#pragma unroll
      for (int i = 0; i < 8; ++i) mx[i] = fmaxf(mx[i], sc[nt][i]);
#pragma unroll
    for (int i = 0; i < 8; ++i) mx[i] = half16_max(mx[i]);

    v8f mnew, corr;
#pragma unroll
    for (int i = 0; i < 8; ++i) {
      mnew[i] = fmaxf(m_acc[i], mx[i]);
      corr[i] = __expf(m_acc[i] - mnew[i]);
    }

    v8f psum = zero8();
#pragma unroll
    for (int nt = 0; nt < 4; ++nt)
#pragma unroll
      for (int i = 0; i < 8; ++i) {
        float e = __expf(sc[nt][i] - mnew[i]);
        sc[nt][i] = e;
        psum[i] += e;
      }
#pragma unroll
    for (int i = 0; i < 8; ++i) psum[i] = half16_sum(psum[i]);

#pragma unroll
    for (int i = 0; i < 8; ++i) {
      l_acc[i] = l_acc[i] * corr[i] + psum[i];
      m_acc[i] = mnew[i];
    }
#pragma unroll
    for (int nt = 0; nt < 4; ++nt)
#pragma unroll
      for (int i = 0; i < 8; ++i) o[nt][i] *= corr[i];

    // P: C/D layout -> A layout via wave-private LDS. Hardware keeps same-wave
    // DS ops in order; the compiler fence stops per-thread reordering (the
    // cross-lane store/load pairs don't alias within one thread's view).
#pragma unroll
    for (int nt = 0; nt < 4; ++nt)
#pragma unroll
      for (int r = 0; r < 8; ++r)
        sP[wave][r + 8 * half][nt * 16 + frow] = (bf16_t)sc[nt][r];

    asm volatile("" ::: "memory");

    v16bf pf0 = frag_from(&sP[wave][frow][fk]);
    v16bf pf1 = frag_from(&sP[wave][frow][32 + fk]);

    // O += P @ V : B operand from V^T [dh][key]
#pragma unroll
    for (int nt = 0; nt < 4; ++nt) {
      v16bf b0 = frag_from(&sVt[nt * 16 + frow][fk]);
      o[nt] = WMMA_BF16(pf0, b0, o[nt]);
      v16bf b1 = frag_from(&sVt[nt * 16 + frow][32 + fk]);
      o[nt] = WMMA_BF16(pf1, b1, o[nt]);
    }
  }

  // Normalize and write [B,T,C] bf16
#pragma unroll
  for (int nt = 0; nt < 4; ++nt)
#pragma unroll
    for (int r = 0; r < 8; ++r) {
      int row = chunk * CSIZE + qt * 128 + wave * 16 + r + 8 * half;
      outb[((size_t)b * TLEN + row) * CDIM + h * DH + nt * 16 + frow] =
          (bf16_t)(o[nt][r] / l_acc[r]);
    }
}

// ---------------------------------------------------------------------------
// Launcher
// ---------------------------------------------------------------------------
extern "C" void kernel_launch(void* const* d_in, const int* in_sizes, int n_in,
                              void* d_out, int out_size, void* d_ws, size_t ws_size,
                              hipStream_t stream) {
  const float* x    = (const float*)d_in[0];
  const float* Wqkv = (const float*)d_in[1];
  const float* Wout = (const float*)d_in[2];
  const float* bout = (const float*)d_in[3];
  const float* Wl1  = (const float*)d_in[4];
  const float* bl1  = (const float*)d_in[5];
  const float* Wl2  = (const float*)d_in[6];
  const float* bl2  = (const float*)d_in[7];
  float* out = (float*)d_out;

  char* ws = (char*)d_ws;
  bf16_t* xb    = (bf16_t*)(ws + 0);           // 16384*1024*2   = 33554432
  bf16_t* wqkvb = (bf16_t*)(ws + 33554432);    // 3072*1024*2    = 6291456
  bf16_t* woutb = (bf16_t*)(ws + 39845888);    // 1024*1024*2    = 2097152
  bf16_t* qkvb  = (bf16_t*)(ws + 41943040);    // 16384*3072*2   = 100663296
  bf16_t* attnb = (bf16_t*)(ws + 142606336);   // 16384*1024*2   = 33554432
  float*  lamp  = (float*)(ws + 176160768);    // 16*4096*4      = 262144

  const int nx = MROWS * CDIM;           // 16777216
  const int nwq = 3 * CDIM * CDIM;       // 3145728
  const int nwo = CDIM * CDIM;           // 1048576

  cvt_f32_bf16_kernel<<<(nx + 255) / 256, 256, 0, stream>>>(x, xb, nx);
  cvt_f32_bf16_kernel<<<(nwq + 255) / 256, 256, 0, stream>>>(Wqkv, wqkvb, nwq);
  cvt_f32_bf16_kernel<<<(nwo + 255) / 256, 256, 0, stream>>>(Wout, woutb, nwo);
  lambda_kernel<<<TLEN / 256, 256, 0, stream>>>(Wl1, bl1, Wl2, bl2, lamp);

  // qkv = x @ Wqkv^T : M=16384, N=3072, K=1024
  gemm_bf16_kernel<false><<<dim3(MROWS / 128, 3 * CDIM / 128), 256, 0, stream>>>(
      xb, wqkvb, qkvb, nullptr, MROWS, 3 * CDIM, CDIM);

  // chunked gated attention: B*H*nc*(cs/128) = 4*16*8*4 = 2048 blocks
  attn_kernel<<<BATCH * HEADS * NCHUNK * (CSIZE / 128), 256, 0, stream>>>(
      qkvb, lamp, attnb);

  // out = attn @ Wout^T + bout : M=16384, N=1024, K=1024
  gemm_bf16_kernel<true><<<dim3(MROWS / 128, CDIM / 128), 256, 0, stream>>>(
      attnb, woutb, out, bout, MROWS, CDIM, CDIM);
}
